// Pillar_Self_Attention_90950227460842
// MI455X (gfx1250) — matrix operand order repaired and checked by reading, not compiled, not run on hardware
//
#include <hip/hip_runtime.h>
#include <hip/hip_bf16.h>

typedef __attribute__((ext_vector_type(16))) _Float16 v16h;
typedef __attribute__((ext_vector_type(8)))  float    v8f;

// Problem constants (fixed by the reference)
#define NB  8
#define CC_ 256
#define CQ  64
#define PP  4096

// ---------------------------------------------------------------------------
// Kernel 1: 1x1-conv projections q = Wq x + bq, k = Wk x + bk, v = Wv x + bv.
// q,k stored TRANSPOSED (position-major, dim-contiguous) in f16 for WMMA
// A/B fragment loads; v stored CHANNEL-major (like x) so the attention
// kernel can stage a [c][q] LDS tile with contiguous-q rows.
// Grid: NB * (PP/64) blocks, 256 threads.
// ---------------------------------------------------------------------------
__global__ __launch_bounds__(256)
void proj_kernel(const float* __restrict__ x,
                 const float* __restrict__ Wq, const float* __restrict__ bq,
                 const float* __restrict__ Wk, const float* __restrict__ bk,
                 const float* __restrict__ Wv, const float* __restrict__ bv,
                 _Float16* __restrict__ qT, _Float16* __restrict__ kT,
                 _Float16* __restrict__ vC)
{
    __shared__ _Float16 xs[CC_][64];            // 32 KB: x tile, f16 staged

    const int n     = blockIdx.x >> 6;
    const int pbase = (blockIdx.x & 63) << 6;
    const int t     = threadIdx.x;

    const float* xn = x + (size_t)n * CC_ * PP + pbase;
    for (int l = t; l < CC_ * 64; l += 256) {
        int c = l >> 6, pl = l & 63;
        xs[c][pl] = (_Float16)xn[(size_t)c * PP + pl];
    }
    __syncthreads();

    const int pl = t & 63;          // position within tile
    const int g  = t >> 6;          // wave-pair uniform output group 0..3

    float accQ[16], accK[16], accV[64];
#pragma unroll
    for (int i = 0; i < 16; ++i) { accQ[i] = 0.f; accK[i] = 0.f; }
#pragma unroll
    for (int i = 0; i < 64; ++i) accV[i] = 0.f;

    for (int c = 0; c < CC_; ++c) {
        const float xv = (float)xs[c][pl];
#pragma unroll
        for (int i = 0; i < 16; ++i)
            accQ[i] = fmaf(Wq[(g * 16 + i) * CC_ + c], xv, accQ[i]);
#pragma unroll
        for (int i = 0; i < 16; ++i)
            accK[i] = fmaf(Wk[(g * 16 + i) * CC_ + c], xv, accK[i]);
#pragma unroll
        for (int i = 0; i < 64; ++i)
            accV[i] = fmaf(Wv[(g * 64 + i) * CC_ + c], xv, accV[i]);
    }

    const size_t prow = (size_t)(n * PP + pbase + pl);
#pragma unroll
    for (int i = 0; i < 16; ++i)
        qT[prow * CQ + g * 16 + i] = (_Float16)(accQ[i] + bq[g * 16 + i]);
#pragma unroll
    for (int i = 0; i < 16; ++i)
        kT[prow * CQ + g * 16 + i] = (_Float16)(accK[i] + bk[g * 16 + i]);
#pragma unroll
    for (int i = 0; i < 64; ++i)     // channel-major: coalesced over positions
        vC[((size_t)(n * CC_) + g * 64 + i) * PP + pbase + pl] =
            (_Float16)(accV[i] + bv[g * 64 + i]);
}

// ---------------------------------------------------------------------------
// Async global -> LDS staging of one (K 8KB + V 32KB) column tile.
// Uses CDNA5 GLOBAL_LOAD_ASYNC_TO_LDS_B128 (ASYNCcnt-tracked DMA): 16B per
// lane per issue, no VGPR staging, overlaps with WMMA compute.
// 256 threads: 2 issues for K, 8 for V.
// ---------------------------------------------------------------------------
__device__ __forceinline__ void async_tile_load(
    int tid, int n, int colBase,
    const _Float16* __restrict__ kT, const _Float16* __restrict__ vC,
    uint32_t kOff, uint32_t vOff)
{
    // K tile: 64 cols x 64 d halves = 512 x 16B
#pragma unroll
    for (int i = 0; i < 2; ++i) {
        int l = tid + i * 256;
        int col = l >> 3, u = l & 7;
        uint32_t lo = kOff + (uint32_t)(col * 128 + u * 16);
        const _Float16* g = kT + (size_t)(n * PP + colBase + col) * CQ + u * 8;
        asm volatile("global_load_async_to_lds_b128 %0, %1, off"
                     :: "v"(lo), "v"(g) : "memory");
    }
    // V tile (channel-major): 256 c x 64 q halves = 2048 x 16B
#pragma unroll
    for (int i = 0; i < 8; ++i) {
        int l = tid + i * 256;
        int c = l >> 3, u = l & 7;
        uint32_t lo = vOff + (uint32_t)(c * 128 + u * 16);
        const _Float16* g = vC + ((size_t)(n * CC_) + c) * PP + colBase + u * 8;
        asm volatile("global_load_async_to_lds_b128 %0, %1, off"
                     :: "v"(lo), "v"(g) : "memory");
    }
}

// ---------------------------------------------------------------------------
// Kernel 2: flash-style attention with double-buffered async tiles.
// Block = 256 threads = 8 wave32. Wave w: row band (w>>1) of 16 Q rows,
// channel half (w&1) of 128 output channels. The S=Q.K^T GEMM + softmax is
// duplicated per channel-half pair; the big P.V accumulator is halved so the
// kernel stays under 256 VGPRs (no s_set_vgpr_msb, 2 waves/SIMD).
// Dynamic LDS: K[2][64][64] | V[2][256][64] | pScr[8][16][64] = 98304 B.
// ---------------------------------------------------------------------------
#define LDS_K0    0u
#define LDS_K1    8192u
#define LDS_V0    16384u
#define LDS_V1    49152u
#define LDS_PSCR  81920u
#define LDS_TOTAL 98304u

__global__ __launch_bounds__(256)
void attn_kernel(const float* __restrict__ x,
                 const _Float16* __restrict__ qT,
                 const _Float16* __restrict__ kT,
                 const _Float16* __restrict__ vC,
                 const float* __restrict__ gamma,
                 float* __restrict__ out)
{
    extern __shared__ char smem[];
    // ISA aperture rule: low 32 bits of a generic LDS address are the
    // wave-relative LDS byte offset -> usable as async-load VDST operand.
    const uint32_t ldsBase = (uint32_t)(size_t)(void*)smem;

    const int n       = blockIdx.x >> 6;
    const int rowTile = (blockIdx.x & 63) << 6;
    const int tid     = threadIdx.x;
    const int wave    = tid >> 5;
    const int lane    = tid & 31;
    const int band    = wave >> 1;              // 0..3: 16-row band
    const int half    = wave & 1;               // 0..1: 128-channel half
    const int rowBase = rowTile + band * 16;
    const int cBase   = half * 128;

    const int arow = lane & 15;                 // A-frag row (M)
    const int ksel = lane >> 4;                 // lane-half selector
    const int bcol = lane & 15;                 // B/D-frag column (N)
    const int kb   = ksel * 16;                 // B-frag K base for this half

    _Float16* pScrW = (_Float16*)(smem + LDS_PSCR) + wave * (16 * 64);

    // --- Q A-fragments (16x32 f16, contiguous k = 16*(lane/16) + s), resident all kernel
    v16h qa[2];
    {
        const _Float16* qrow = qT + (size_t)(n * PP + rowBase + arow) * CQ;
#pragma unroll
        for (int f = 0; f < 2; ++f)
#pragma unroll
            for (int j = 0; j < 8; ++j) {
                int kk = f * 32 + ksel * 16 + 2 * j;
                qa[f][2 * j]     = qrow[kk];
                qa[f][2 * j + 1] = qrow[kk + 1];
            }
    }

    const v8f vzero = {};
    v8f O[8];                                   // 16 rows x 128 channels
#pragma unroll
    for (int cc = 0; cc < 8; ++cc) O[cc] = vzero;

    float mRun[8], lRun[8];
#pragma unroll
    for (int j = 0; j < 8; ++j) { mRun[j] = -__builtin_inff(); lRun[j] = 0.f; }

    // Prologue: kick off async DMA for tile 0 into buffer 0
    async_tile_load(tid, n, 0, kT, vC, ldsBase + LDS_K0, ldsBase + LDS_V0);

    for (int jt = 0; jt < 64; ++jt) {
        const int cur = jt & 1;

        // Own-wave async completion, then block-wide visibility
        asm volatile("s_wait_asynccnt 0x0" ::: "memory");
        __syncthreads();

        // Prefetch next tile into the other buffer (overlaps with compute)
        if (jt < 63)
            async_tile_load(tid, n, (jt + 1) * 64, kT, vC,
                            ldsBase + (cur ? LDS_K0 : LDS_K1),
                            ldsBase + (cur ? LDS_V0 : LDS_V1));

        const _Float16* kTile  = (const _Float16*)(smem + (cur ? LDS_K1 : LDS_K0));
        const _Float16* vTileT = (const _Float16*)(smem + (cur ? LDS_V1 : LDS_V0));

        // ---- S = Q . K^T : four 16x16 tiles, K-dim 64 = 2 x 32
        v8f S[4];
#pragma unroll
        for (int ct = 0; ct < 4; ++ct) {
            v8f s = vzero;
#pragma unroll
            for (int f = 0; f < 2; ++f) {
                v16h b;
#pragma unroll
                for (int j = 0; j < 8; ++j) {
                    int kk = f * 32 + kb + 2 * j;
                    b[2 * j]     = kTile[(ct * 16 + bcol) * CQ + kk];
                    b[2 * j + 1] = kTile[(ct * 16 + bcol) * CQ + kk + 1];
                }
                s = __builtin_amdgcn_wmma_f32_16x16x32_f16(
                        false, qa[f], false, b, (short)0, s, false, false);
            }
            S[ct] = s;
        }

        // ---- streaming softmax: rows j (lanes 0-15: M=j, lanes 16-31: M=j+8)
#pragma unroll
        for (int j = 0; j < 8; ++j) {
            float mt = fmaxf(fmaxf(S[0][j], S[1][j]), fmaxf(S[2][j], S[3][j]));
            mt = fmaxf(mt, __shfl_xor(mt, 1, 32));
            mt = fmaxf(mt, __shfl_xor(mt, 2, 32));
            mt = fmaxf(mt, __shfl_xor(mt, 4, 32));
            mt = fmaxf(mt, __shfl_xor(mt, 8, 32));
            const float mN   = fmaxf(mRun[j], mt);
            const float corr = __expf(mRun[j] - mN);
            mRun[j] = mN;

            float rs = 0.f;
#pragma unroll
            for (int ct = 0; ct < 4; ++ct) {
                float e = __expf(S[ct][j] - mN);
                rs += e;
                pScrW[(j + ksel * 8) * 64 + ct * 16 + bcol] = (_Float16)e;
            }
            rs += __shfl_xor(rs, 1, 32);
            rs += __shfl_xor(rs, 2, 32);
            rs += __shfl_xor(rs, 4, 32);
            rs += __shfl_xor(rs, 8, 32);
            lRun[j] = lRun[j] * corr + rs;
#pragma unroll
            for (int cc = 0; cc < 8; ++cc) O[cc][j] *= corr;
        }

        // intra-wave LDS RAW: explicit CDNA5 split-counter wait
        asm volatile("s_wait_dscnt 0x0" ::: "memory");

        // ---- reload P as A-fragments (D-layout -> A-layout transpose via LDS)
        v16h pa[2];
#pragma unroll
        for (int f = 0; f < 2; ++f)
#pragma unroll
            for (int j = 0; j < 8; ++j) {
                int kk = f * 32 + ksel * 16 + 2 * j;
                pa[f][2 * j]     = pScrW[arow * 64 + kk];
                pa[f][2 * j + 1] = pScrW[arow * 64 + kk + 1];
            }

        // ---- O += P . V : this wave's 8 column chunks, K-dim 64 = 2 x 32
        //      B[k=q][n=c] = vTileT[c][q]: 16 contiguous halves per lane
#pragma unroll
        for (int cc = 0; cc < 8; ++cc) {
            v8f acc = O[cc];
#pragma unroll
            for (int f = 0; f < 2; ++f) {
                v16h b;
#pragma unroll
                for (int j = 0; j < 8; ++j) {
                    int kk = f * 32 + kb + 2 * j;
                    b[2 * j]     = vTileT[(cBase + cc * 16 + bcol) * 64 + kk];
                    b[2 * j + 1] = vTileT[(cBase + cc * 16 + bcol) * 64 + kk + 1];
                }
                acc = __builtin_amdgcn_wmma_f32_16x16x32_f16(
                          false, pa[f], false, b, (short)0, acc, false, false);
            }
            O[cc] = acc;
        }
    }

    // ---- epilogue: out = gamma * (O / l) + x, transposed through per-wave
    //      LDS scratch so global accesses run with lanes along p (coalesced).
    const float gm = gamma[0];
    float inv[8];
#pragma unroll
    for (int j = 0; j < 8; ++j) inv[j] = 1.f / lRun[j];

    float* scrF = (float*)pScrW;                // 16x16 f32 tile: [c_local][p_local]
    const int plo   = lane & 15;
    const int chalf = (lane >> 4) * 8;

#pragma unroll
    for (int cc = 0; cc < 8; ++cc) {
        asm volatile("s_wait_dscnt 0x0" ::: "memory");  // prior reads done
#pragma unroll
        for (int j = 0; j < 8; ++j)                     // write (c_local=bcol, p_local=row)
            scrF[bcol * 16 + (j + ksel * 8)] = O[cc][j] * inv[j];
        asm volatile("s_wait_dscnt 0x0" ::: "memory");  // intra-wave RAW

        const size_t base =
            ((size_t)n * CC_ + cBase + cc * 16) * (size_t)PP + rowBase + plo;
#pragma unroll
        for (int i = 0; i < 8; ++i) {
            const int cl = chalf + i;
            const size_t idx = base + (size_t)cl * PP;
            out[idx] = gm * scrF[cl * 16 + plo] + x[idx];
        }
    }
}

// ---------------------------------------------------------------------------
extern "C" void kernel_launch(void* const* d_in, const int* in_sizes, int n_in,
                              void* d_out, int out_size, void* d_ws, size_t ws_size,
                              hipStream_t stream) {
    (void)in_sizes; (void)n_in; (void)out_size; (void)ws_size;

    const float* x     = (const float*)d_in[0];
    const float* Wq    = (const float*)d_in[1];
    const float* bq    = (const float*)d_in[2];
    const float* Wk    = (const float*)d_in[3];
    const float* bk    = (const float*)d_in[4];
    const float* Wv    = (const float*)d_in[5];
    const float* bv    = (const float*)d_in[6];
    const float* gamma = (const float*)d_in[7];
    float* out         = (float*)d_out;

    // Workspace layout (f16): qT [N*P*CQ] | kT [N*P*CQ] | vC [N*C*P]
    char* ws = (char*)d_ws;
    _Float16* qT = (_Float16*)(ws);
    _Float16* kT = (_Float16*)(ws + (size_t)NB * PP * CQ * 2);
    _Float16* vC = (_Float16*)(ws + (size_t)NB * PP * CQ * 4);

    const int blocks = NB * (PP / 64);   // 512

    // Allow 96 KB dynamic LDS (WGP has 320 KB); attribute set is not a
    // stream op, so it is graph-capture safe and deterministic.
    (void)hipFuncSetAttribute((const void*)attn_kernel,
                              hipFuncAttributeMaxDynamicSharedMemorySize,
                              (int)LDS_TOTAL);

    proj_kernel<<<blocks, 256, 0, stream>>>(x, Wq, bq, Wk, bk, Wv, bv,
                                            qT, kT, vC);
    attn_kernel<<<blocks, 256, LDS_TOTAL, stream>>>(x, qT, kT, vC, gamma, out);
}
